// Attention_498216206551
// MI455X (gfx1250) — matrix operand order repaired and checked by reading, not compiled, not run on hardware
//
#include <hip/hip_runtime.h>

typedef __attribute__((ext_vector_type(16))) _Float16 v16h;
typedef __attribute__((ext_vector_type(8)))  _Float16 v8h;
typedef __attribute__((ext_vector_type(8)))  float    v8f;

#define WMMA_F16(a, b, c) \
  __builtin_amdgcn_wmma_f32_16x16x32_f16(false, (a), false, (b), (short)0, (c), false, false)

constexpr int BATCH = 32;
constexpr int DIM   = 256;
constexpr int HEADS = 8;
constexpr int DH    = 32;
constexpr int NTOK  = 625;   // 25*25
constexpr int NPAD  = 640;   // 40 tiles of 16
constexpr int QKVN  = 768;   // 3*DIM
constexpr float SCALE = 0.17677669529663687f; // 32^-0.5
constexpr float NEGBIG = -1.0e30f;

// ---------------------------------------------------------------------------
// Kernel 0: materialize rel-pos bias once: bias[h][NPAD][NPAD] f32.
// Masked (col >= 625) entries pre-set to -1e30; pad rows use clamped data.
// Removes the dependent rel_idx->rel_emb gather (x32 batch redundancy) from
// the attention hot loop.
// ---------------------------------------------------------------------------
__global__ __launch_bounds__(256) void bias_kernel(
    const float* __restrict__ rel_emb, const int* __restrict__ rel_idx,
    float* __restrict__ bias) {
  const int idx = blockIdx.x * 256 + threadIdx.x;   // h * NPAD * NPAD
  const int col = idx % NPAD;
  const int row = (idx / NPAD) % NPAD;
  const int h   = idx / (NPAD * NPAD);
  if (h >= HEADS) return;
  const int rc = (row < NTOK) ? row : (NTOK - 1);
  const int cc = (col < NTOK) ? col : (NTOK - 1);
  const float v = rel_emb[rel_idx[rc * NTOK + cc] * HEADS + h];
  bias[idx] = (col < NTOK) ? v : NEGBIG;
}

// ---------------------------------------------------------------------------
// Kernel 1: qkv[b,n,3d] = xf[b,n,d] @ w_qkv[d,3d]; split into fp16 Q/K/V
// laid out [b, h, NPAD, DH]; Q pre-scaled; pad rows (n>=625) come out zero.
// One wave -> one 16x16 output tile, K=256 in 8 WMMA steps.
// Loads are branch-free (clamped index + mask multiply) so they clause.
// ---------------------------------------------------------------------------
__global__ __launch_bounds__(128) void qkv_kernel(
    const float* __restrict__ x, const float* __restrict__ w,
    _Float16* __restrict__ Qb, _Float16* __restrict__ Kb, _Float16* __restrict__ Vb) {
  const int lane = threadIdx.x & 31;
  const int wave = threadIdx.x >> 5;
  int tile = blockIdx.x * 4 + wave;            // b(32) * mt(40) * nt(48)
  const int nt = tile % 48; tile /= 48;
  const int mt = tile % 40;
  const int b  = tile / 40;
  if (b >= BATCH) return;

  const int lo = lane & 15, hi = lane >> 4;
  const int i0 = mt * 16, o0 = nt * 16;

  const int   irow    = i0 + lo;                          // A row (token)
  const int   irowc   = (irow < NTOK) ? irow : (NTOK - 1);
  const float rowmask = (irow < NTOK) ? 1.0f : 0.0f;      // zero pad rows
  const float* xrow = x + (size_t)b * DIM * NTOK + irowc; // &x[b][0][irowc]
  const int kbase = hi * 8;

  v8f acc = {};
  for (int kc = 0; kc < 8; ++kc) {
    const int c0 = kc * 32;
    v16h A;                          // xf tile 16x32 (f32 -> f16)
#pragma unroll
    for (int t = 0; t < 16; ++t) {
      const int c = c0 + hi * 16 + t;
      A[t] = (_Float16)(xrow[(size_t)c * NTOK] * rowmask);
    }
    v16h B;                          // w_qkv tile 32x16
#pragma unroll
    for (int t = 0; t < 16; ++t) {
      const int cc = c0 + hi * 16 + t;
      B[t] = (_Float16)w[cc * QKVN + o0 + lo];
    }
    acc = WMMA_F16(A, B, acc);
  }

  const int og     = o0 + lo;        // 0..767
  const int which  = og >> 8;        // 0=q 1=k 2=v
  const int within = og & 255;
  const int h      = within >> 5;
  const int dd     = within & 31;
  _Float16* dst = (which == 0) ? Qb : (which == 1) ? Kb : Vb;
  const float scl = (which == 0) ? SCALE : 1.0f;
  const size_t base = ((size_t)(b * HEADS + h) * NPAD) * DH + dd;
#pragma unroll
  for (int r = 0; r < 8; ++r) {
    const int row = i0 + r + 8 * hi;
    dst[base + (size_t)row * DH] = (_Float16)(acc[r] * scl);
  }
}

// ---------------------------------------------------------------------------
// Kernel 2: fused attention. One wave per (b, h, 16-row tile).
//  - sim strip (16 x 640) via WMMA with precomputed bias loaded into C
//  - softmax stats via LDS scan + 16-lane shfl_xor reductions
//  - O = P @ V via 2 WMMAs per 32-col chunk, normalized by row sum
// Writes fp16 attn-out [b, NPAD, DIM].
// ---------------------------------------------------------------------------
constexpr int STR = 656;  // padded LDS row stride (floats), avoids bank conflicts

__global__ __launch_bounds__(32) void attn_kernel(
    const _Float16* __restrict__ Qb, const _Float16* __restrict__ Kb,
    const _Float16* __restrict__ Vb, const float* __restrict__ bias,
    _Float16* __restrict__ Ob) {
  __shared__ float smem[16 * STR + 16];
  float* rowsum = smem + 16 * STR;

  int tile = blockIdx.x;               // b(32) * h(8) * it(40)
  const int it = tile % 40; tile /= 40;
  const int h  = tile % 8;
  const int b  = tile / 8;

  const int lane = threadIdx.x & 31;
  const int lo = lane & 15, hi = lane >> 4;
  const int i0 = it * 16;
  const size_t bh = (size_t)(b * HEADS + h) * NPAD * DH;
  const _Float16* qb = Qb + bh;
  const _Float16* kb = Kb + bh;
  const _Float16* vb = Vb + bh;
  const float* biasb = bias + (size_t)h * NPAD * NPAD;

  // Q tile: exactly one f16 A operand (16 x 32, K = DH)
  v16h Aq;
  {
    const int m = i0 + lo;
    const int kbase = hi * 16;
    const v8h a0 = *(const v8h*)(qb + m * DH + kbase);
    const v8h a1 = *(const v8h*)(qb + m * DH + 8 + kbase);
#pragma unroll
    for (int t = 0; t < 8; ++t) { Aq[t] = a0[t]; Aq[t + 8] = a1[t]; }
  }

  // Phase 1: scores -> LDS (bias preloaded into the C operand)
  for (int jt = 0; jt < 40; ++jt) {
    const int j0 = jt * 16;
    const v16h Bk = *(const v16h*)(kb + (j0 + lo) * DH + hi * 16);  // K^T 32x16
    v8f C;
#pragma unroll
    for (int r = 0; r < 8; ++r) {
      const int row = i0 + r + 8 * hi;                   // < NPAD always
      C[r] = biasb[(size_t)row * NPAD + j0 + lo];        // coalesced over lanes
    }
    const v8f S = WMMA_F16(Aq, Bk, C);
#pragma unroll
    for (int r = 0; r < 8; ++r)
      smem[(r + 8 * hi) * STR + j0 + lo] = S[r];
  }
  __syncthreads();

  // Phase 2: row max, exp in place, row sum (lane pair (l, l+16) splits cols)
  {
    const int row = lo;
    float m = -3.0e38f;
    for (int t = 0; t < 320; ++t)
      m = fmaxf(m, smem[row * STR + hi * 320 + t]);
    m = fmaxf(m, __shfl_xor(m, 16));
    float s = 0.0f;
    for (int t = 0; t < 320; ++t) {
      const int c = hi * 320 + t;
      const float e = __expf(smem[row * STR + c] - m);
      smem[row * STR + c] = e;
      s += e;
    }
    s += __shfl_xor(s, 16);
    if (hi == 0) rowsum[row] = s;
  }
  __syncthreads();

  // Phase 3: O = P @ V  (dh split into two 16-col halves)
  v8f O0 = {}, O1 = {};
  const int kbase = hi * 8;
  for (int jc = 0; jc < 20; ++jc) {
    const int j0 = jc * 32;
    v16h Ap;                                  // P tile 16x32 gathered from LDS
#pragma unroll
    for (int t = 0; t < 16; ++t) {
      const int j = j0 + hi * 16 + t;
      Ap[t] = (_Float16)smem[lo * STR + j];
    }
    v16h Bv0, Bv1;                            // V chunk 32x32 as two B operands
#pragma unroll
    for (int t = 0; t < 16; ++t) {
      const int kk = j0 + hi * 16 + t;
      Bv0[t] = vb[kk * DH + lo];
      Bv1[t] = vb[kk * DH + 16 + lo];
    }
    O0 = WMMA_F16(Ap, Bv0, O0);
    O1 = WMMA_F16(Ap, Bv1, O1);
  }

  // Normalize and store fp16 attn-out [b, NPAD, DIM]
#pragma unroll
  for (int r = 0; r < 8; ++r) {
    const int lrow = r + 8 * hi;
    const float inv = 1.0f / rowsum[lrow];
    const int row = i0 + lrow;
    const size_t base = ((size_t)b * NPAD + row) * DIM + h * DH;
    Ob[base + lo]      = (_Float16)(O0[r] * inv);
    Ob[base + 16 + lo] = (_Float16)(O1[r] * inv);
  }
}

// ---------------------------------------------------------------------------
// Kernel 3: y[b,n,d] = attn_out @ w_out, stored transposed as [b, d, 25, 25].
// ---------------------------------------------------------------------------
__global__ __launch_bounds__(128) void outproj_kernel(
    const _Float16* __restrict__ Ob, const float* __restrict__ w,
    float* __restrict__ y) {
  const int lane = threadIdx.x & 31;
  const int wave = threadIdx.x >> 5;
  int tile = blockIdx.x * 4 + wave;           // b(32) * mt(40) * nt(16)
  const int nt = tile & 15; tile >>= 4;
  const int mt = tile % 40;
  const int b  = tile / 40;
  if (b >= BATCH) return;

  const int lo = lane & 15, hi = lane >> 4;
  const int i0 = mt * 16, o0 = nt * 16;
  const _Float16* ab = Ob + (size_t)b * NPAD * DIM;

  v8f acc = {};
  const int m = i0 + lo;
  const int kbase = hi * 16;
  for (int kc = 0; kc < 8; ++kc) {
    const int k0 = kc * 32;
    const v8h a0 = *(const v8h*)(ab + m * DIM + k0 + kbase);
    const v8h a1 = *(const v8h*)(ab + m * DIM + k0 + 8 + kbase);
    v16h A;
#pragma unroll
    for (int t = 0; t < 8; ++t) { A[t] = a0[t]; A[t + 8] = a1[t]; }
    v16h B;
#pragma unroll
    for (int t = 0; t < 16; ++t) {
      const int kk = k0 + hi * 16 + t;
      B[t] = (_Float16)w[kk * DIM + o0 + lo];
    }
    acc = WMMA_F16(A, B, acc);
  }
#pragma unroll
  for (int r = 0; r < 8; ++r) {
    const int row = i0 + r + 8 * hi;
    if (row < NTOK) {
      const int c = o0 + lo;
      y[((size_t)b * DIM + c) * NTOK + row] = acc[r];
    }
  }
}

// ---------------------------------------------------------------------------
extern "C" void kernel_launch(void* const* d_in, const int* in_sizes, int n_in,
                              void* d_out, int out_size, void* d_ws, size_t ws_size,
                              hipStream_t stream) {
  (void)in_sizes; (void)n_in; (void)out_size; (void)ws_size;
  const float* x       = (const float*)d_in[0];
  const float* w_qkv   = (const float*)d_in[1];
  const float* rel_emb = (const float*)d_in[2];
  const float* w_out   = (const float*)d_in[3];
  const int*   rel_idx = (const int*)d_in[4];

  char* ws = (char*)d_ws;
  const size_t sz = (size_t)BATCH * HEADS * NPAD * DH * sizeof(_Float16); // 10 MB
  _Float16* Qb   = (_Float16*)(ws);
  _Float16* Kb   = (_Float16*)(ws + sz);
  _Float16* Vb   = (_Float16*)(ws + 2 * sz);
  _Float16* Ob   = (_Float16*)(ws + 3 * sz);
  float*    bias = (float*)   (ws + 4 * sz);   // 8*640*640*4 = 13.1 MB

  // 8*640*640 = 3,276,800 elements
  bias_kernel<<<12800, 256, 0, stream>>>(rel_emb, rel_idx, bias);
  // 32*40*48 = 61440 tiles, 4 waves/block
  qkv_kernel<<<15360, 128, 0, stream>>>(x, w_qkv, Qb, Kb, Vb);
  // 32*8*40 = 10240 (b, h, row-tile) workgroups, one wave each
  attn_kernel<<<10240, 32, 0, stream>>>(Qb, Kb, Vb, bias, Ob);
  // 32*40*16 = 20480 tiles, 4 waves/block
  outproj_kernel<<<5120, 128, 0, stream>>>(Ob, w_out, (float*)d_out);
}
